// PointNextEncoder_74852690035267
// MI455X (gfx1250) — matrix operand order repaired and checked by reading, not compiled, not run on hardware
//
#include <hip/hip_runtime.h>
#include <hip/hip_bf16.h>
#include <math.h>

typedef __attribute__((ext_vector_type(16))) _Float16 v16h;
typedef __attribute__((ext_vector_type(8)))  _Float16 v8h;
typedef __attribute__((ext_vector_type(8)))  float    v8f;

__device__ __forceinline__ float gelu_exact(float x) {
  return 0.5f * x * (1.0f + erff(x * 0.7071067811865476f));
}

__device__ __forceinline__ void ln_row(float* rowp, int C, const float* bias,
                                       const float* gamma, const float* beta, int act) {
  float mu = 0.f;
  for (int c = 0; c < C; ++c) { float v = rowp[c] + bias[c]; rowp[c] = v; mu += v; }
  mu /= (float)C;
  float var = 0.f;
  for (int c = 0; c < C; ++c) { float d = rowp[c] - mu; var += d * d; }
  var /= (float)C;
  float inv = rsqrtf(var + 1e-5f);
  for (int c = 0; c < C; ++c) {
    float y = (rowp[c] - mu) * inv * gamma[c] + beta[c];
    rowp[c] = act ? gelu_exact(y) : y;
  }
}

__device__ __forceinline__ void ln_row_to_h(float* raw, _Float16* out, int C,
                                            const float* bias, const float* gamma,
                                            const float* beta, int act) {
  float mu = 0.f;
  for (int c = 0; c < C; ++c) { float v = raw[c] + bias[c]; raw[c] = v; mu += v; }
  mu /= (float)C;
  float var = 0.f;
  for (int c = 0; c < C; ++c) { float d = raw[c] - mu; var += d * d; }
  var /= (float)C;
  float inv = rsqrtf(var + 1e-5f);
  for (int c = 0; c < C; ++c) {
    float y = (raw[c] - mu) * inv * gamma[c] + beta[c];
    out[c] = (_Float16)(act ? gelu_exact(y) : y);
  }
}

// A fragment: two contiguous 16B runs from LDS -> two ds_load_b128.
__device__ __forceinline__ v16h load_a_lds(const _Float16* sX, int ld, int row,
                                           int kbase, int lane) {
  const int khalf = (lane < 16) ? 0 : 16;
  const _Float16* rp = sX + (size_t)row * ld + kbase + khalf;
  v8h lo = *(const v8h*)rp;
  v8h hi = *(const v8h*)(rp + 8);
  return __builtin_shufflevector(lo, hi, 0,1,2,3,4,5,6,7,8,9,10,11,12,13,14,15);
}

// B fragment: 16 contiguous K values at column ncol from N-major Wt[Cout][Cinp].
__device__ __forceinline__ v16h load_b_t(const _Float16* Wt, int Cinp, int ncol,
                                         int kbase, int lane) {
  const int kh = (lane < 16) ? 0 : 16;
  return *(const v16h*)(Wt + (size_t)ncol * Cinp + kbase + kh);
}

__device__ __forceinline__ void store_d(float* dst, int ldc, int ncol, int lane, v8f acc) {
#pragma unroll
  for (int j = 0; j < 8; ++j) {
    int r = (lane < 16) ? j : (j + 8);
    dst[(size_t)r * ldc + ncol] = acc[j];
  }
}

// ---------------------------------------------------------------------------
__global__ void k_pmlp(const float* __restrict__ X, int R, int Cin,
                       const _Float16* __restrict__ Wt, int Cinp, int Cout,
                       const float* __restrict__ bias, const float* __restrict__ gamma,
                       const float* __restrict__ betav, float* __restrict__ Y, int act) {
  extern __shared__ char smem[];
  _Float16* sX = (_Float16*)smem;
  float*    sC = (float*)(smem + (size_t)64 * Cinp * sizeof(_Float16));
  const int lane = threadIdx.x & 31;
  const int wave = threadIdx.x >> 5;
  const int rowblk = blockIdx.x * 64;

  for (int i = threadIdx.x; i < 64 * Cinp; i += blockDim.x) {
    int r = i / Cinp, c = i % Cinp;
    int gr = rowblk + r;
    float v = (gr < R && c < Cin) ? X[(size_t)gr * Cin + c] : 0.f;
    sX[i] = (_Float16)v;
  }
  __syncthreads();

  const int arow = wave * 16 + (lane & 15);
  for (int nt = 0; nt < (Cout >> 4); ++nt) {
    v8f acc = {};
    const int ncol = nt * 16 + (lane & 15);
    for (int kt = 0; kt < Cinp; kt += 32) {
      __builtin_prefetch(Wt + (size_t)ncol * Cinp + kt + 32, 0, 1);
      v16h a  = load_a_lds(sX, Cinp, arow, kt, lane);
      v16h bb = load_b_t(Wt, Cinp, ncol, kt, lane);
      acc = __builtin_amdgcn_wmma_f32_16x16x32_f16(false, a, false, bb,
                                                   (short)0, acc, false, false);
    }
    store_d(&sC[(size_t)(wave * 16) * Cout], Cout, ncol, lane, acc);
  }
  __syncthreads();

  if (threadIdx.x < 64) {
    int gr = rowblk + (int)threadIdx.x;
    if (gr < R) {
      float* rowp = &sC[(size_t)threadIdx.x * Cout];
      ln_row(rowp, Cout, bias, gamma, betav, act);
      float* yo = Y + (size_t)gr * Cout;
      for (int c = 0; c < Cout; ++c) yo[c] = rowp[c];
    }
  }
}

// ---------------------------------------------------------------------------
__global__ void k_fps(const float* __restrict__ pos, int N, int m, int* __restrict__ fidx) {
  extern __shared__ float smf[];
  float* sx = smf; float* sy = smf + N; float* sz = smf + 2 * N;
  __shared__ float rv[256];
  __shared__ int   ri[256];
  __shared__ int   s_last;
  const int tid = threadIdx.x;
  const float* p = pos + (size_t)blockIdx.x * N * 3;
  int* fo = fidx + (size_t)blockIdx.x * m;
  for (int i = tid; i < N; i += 256) { sx[i] = p[3*i]; sy[i] = p[3*i+1]; sz[i] = p[3*i+2]; }
  float mind[16];
#pragma unroll
  for (int j = 0; j < 16; ++j) mind[j] = 1e10f;
  if (tid == 0) { fo[0] = 0; s_last = 0; }
  __syncthreads();
  for (int it = 1; it < m; ++it) {
    const int last = s_last;
    const float lx = sx[last], ly = sy[last], lz = sz[last];
    float best = -1.f; int bi = 0;
    int jj = 0;
    for (int i = tid; i < N; i += 256, ++jj) {
      float dx = sx[i]-lx, dy = sy[i]-ly, dz = sz[i]-lz;
      float d = dx*dx + dy*dy + dz*dz;
      float md = fminf(mind[jj], d);
      mind[jj] = md;
      if (md > best) { best = md; bi = i; }
    }
    rv[tid] = best; ri[tid] = bi;
    __syncthreads();
    for (int s = 128; s > 0; s >>= 1) {
      if (tid < s) {
        if (rv[tid+s] > rv[tid] || (rv[tid+s] == rv[tid] && ri[tid+s] < ri[tid])) {
          rv[tid] = rv[tid+s]; ri[tid] = ri[tid+s];
        }
      }
      __syncthreads();
    }
    if (tid == 0) { fo[it] = ri[0]; s_last = ri[0]; }
    __syncthreads();
  }
}

__global__ void k_gather_pos(const float* __restrict__ pos, int N, const int* __restrict__ fidx,
                             int m, int Bm, float* __restrict__ npos) {
  int t = blockIdx.x * blockDim.x + threadIdx.x;
  if (t >= Bm) return;
  int b = t / m;
  int idx = fidx[t];
  const float* s = pos + ((size_t)b * N + idx) * 3;
  float* d = npos + (size_t)t * 3;
  d[0] = s[0]; d[1] = s[1]; d[2] = s[2];
}

__global__ void k_ballq(const float* __restrict__ pos, int N, const float* __restrict__ npos,
                        int m, int k, float r2, int Bm, int* __restrict__ nidx) {
  int t = blockIdx.x * blockDim.x + threadIdx.x;
  if (t >= Bm) return;
  int b = t / m;
  const float qx = npos[(size_t)t*3], qy = npos[(size_t)t*3+1], qz = npos[(size_t)t*3+2];
  const float* p = pos + (size_t)b * N * 3;
  int buf[24];
  int cnt = 0;
  for (int j = 0; j < N && cnt < k; ++j) {
    float dx = p[3*j]-qx, dy = p[3*j+1]-qy, dz = p[3*j+2]-qz;
    if (dx*dx + dy*dy + dz*dz < r2) buf[cnt++] = j;
  }
  int fill = (cnt > 0) ? buf[0] : 0;
  int* o = nidx + (size_t)t * k;
  for (int j = 0; j < k; ++j) o[j] = (j < cnt) ? buf[j] : fill;
}

__global__ void k_gather_feat(const float* __restrict__ feat, int N, int Cf,
                              const int* __restrict__ fidx, int m, int Bm,
                              float* __restrict__ out) {
  int t = blockIdx.x * blockDim.x + threadIdx.x;
  if (t >= Bm) return;
  int b = t / m;
  int idx = fidx[t];
  const float* s = feat + ((size_t)b * N + idx) * Cf;
  float* d = out + (size_t)t * Cf;
  for (int c = 0; c < Cf; ++c) d[c] = s[c];
}

// ---------------------------------------------------------------------------
__global__ void k_group_mlp(const float* __restrict__ pos, const float* __restrict__ feat,
                            const float* __restrict__ npos, const int* __restrict__ nidx,
                            int N, int m, int k, int Cf, int Cgp, int C0, int C1, int QB,
                            const float* __restrict__ rpeW, const float* __restrict__ rpeB,
                            const float* __restrict__ rpeG, const float* __restrict__ rpeBeta,
                            const _Float16* __restrict__ W0t, const float* __restrict__ b0,
                            const float* __restrict__ g0, const float* __restrict__ be0,
                            const _Float16* __restrict__ W1t, const float* __restrict__ b1,
                            const float* __restrict__ g1, const float* __restrict__ be1,
                            float* __restrict__ hmax) {
  extern __shared__ char smem[];
  const int rows = QB * k;
  _Float16* sG   = (_Float16*)smem;
  float*    sH1f = (float*)(smem + (size_t)rows * Cgp * 2);
  _Float16* sH1h = (_Float16*)((char*)sH1f + (size_t)rows * C0 * 4);
  float*    sH2  = (float*)((char*)sH1h + (size_t)rows * C0 * 2);
  const int nblk = m / QB;
  const int b  = blockIdx.x / nblk;
  const int q0 = (blockIdx.x % nblk) * QB;
  const int lane = threadIdx.x & 31;
  const int wave = threadIdx.x >> 5;
  const int Cg = 3 + Cf + 16;

  for (int r = threadIdx.x; r < rows; r += blockDim.x) {
    const int q = q0 + r / k;
    const int j = r % k;
    const int nb = nidx[((size_t)b * m + q) * k + j];
    const float cx = npos[((size_t)b*m+q)*3+0];
    const float cy = npos[((size_t)b*m+q)*3+1];
    const float cz = npos[((size_t)b*m+q)*3+2];
    const float gx = pos[((size_t)b*N+nb)*3+0] - cx;
    const float gy = pos[((size_t)b*N+nb)*3+1] - cy;
    const float gz = pos[((size_t)b*N+nb)*3+2] - cz;
    _Float16* row = &sG[(size_t)r * Cgp];
    row[0] = (_Float16)gx; row[1] = (_Float16)gy; row[2] = (_Float16)gz;
    const float* fp = feat + ((size_t)b*N + nb) * Cf;
    for (int c = 0; c < Cf; ++c) row[3 + c] = (_Float16)fp[c];
    float rel[16]; float mu = 0.f;
#pragma unroll
    for (int c = 0; c < 16; ++c) {
      float v = rpeW[c]*gx + rpeW[16+c]*gy + rpeW[32+c]*gz + rpeB[c];
      rel[c] = v; mu += v;
    }
    mu *= (1.f/16.f);
    float var = 0.f;
#pragma unroll
    for (int c = 0; c < 16; ++c) { float d = rel[c] - mu; var += d * d; }
    var *= (1.f/16.f);
    float inv = rsqrtf(var + 1e-5f);
#pragma unroll
    for (int c = 0; c < 16; ++c)
      row[3 + Cf + c] = (_Float16)gelu_exact((rel[c]-mu)*inv*rpeG[c] + rpeBeta[c]);
    for (int c = Cg; c < Cgp; ++c) row[c] = (_Float16)0.f;
  }
  __syncthreads();

  const int mtiles = rows >> 4;
  for (int t = wave; t < mtiles; t += 4) {
    const int arow = t * 16 + (lane & 15);
    for (int nt = 0; nt < (C0 >> 4); ++nt) {
      v8f acc = {};
      const int ncol = nt * 16 + (lane & 15);
      for (int kt = 0; kt < Cgp; kt += 32) {
        v16h a  = load_a_lds(sG, Cgp, arow, kt, lane);
        v16h bb = load_b_t(W0t, Cgp, ncol, kt, lane);
        acc = __builtin_amdgcn_wmma_f32_16x16x32_f16(false, a, false, bb,
                                                     (short)0, acc, false, false);
      }
      store_d(&sH1f[(size_t)(t * 16) * C0], C0, ncol, lane, acc);
    }
  }
  __syncthreads();
  for (int r = threadIdx.x; r < rows; r += blockDim.x)
    ln_row_to_h(&sH1f[(size_t)r * C0], &sH1h[(size_t)r * C0], C0, b0, g0, be0, 1);
  __syncthreads();

  for (int t = wave; t < mtiles; t += 4) {
    const int arow = t * 16 + (lane & 15);
    for (int nt = 0; nt < (C1 >> 4); ++nt) {
      v8f acc = {};
      const int ncol = nt * 16 + (lane & 15);
      for (int kt = 0; kt < C0; kt += 32) {
        v16h a  = load_a_lds(sH1h, C0, arow, kt, lane);
        v16h bb = load_b_t(W1t, C0, ncol, kt, lane);
        acc = __builtin_amdgcn_wmma_f32_16x16x32_f16(false, a, false, bb,
                                                     (short)0, acc, false, false);
      }
      store_d(&sH2[(size_t)(t * 16) * C1], C1, ncol, lane, acc);
    }
  }
  __syncthreads();
  for (int r = threadIdx.x; r < rows; r += blockDim.x)
    ln_row(&sH2[(size_t)r * C1], C1, b1, g1, be1, 0);
  __syncthreads();

  for (int t = threadIdx.x; t < QB * C1; t += blockDim.x) {
    const int q = t / C1, c = t % C1;
    float mx = -3.0e38f;
    for (int j = 0; j < k; ++j) mx = fmaxf(mx, sH2[(size_t)(q * k + j) * C1 + c]);
    hmax[((size_t)b * m + q0 + q) * C1 + c] = mx;
  }
}

__global__ void k_combine(const float* __restrict__ h, const float* __restrict__ idn,
                          float* __restrict__ o, size_t n) {
  size_t t = (size_t)blockIdx.x * blockDim.x + threadIdx.x;
  if (t < n) o[t] = gelu_exact(h[t] + idn[t]);
}

__global__ void k_pool_max(const float* __restrict__ feat, int m, int C, int BC,
                           float* __restrict__ out) {
  int t = blockIdx.x * blockDim.x + threadIdx.x;
  if (t >= BC) return;
  int b = t / C, c = t % C;
  float mx = -3.0e38f;
  const float* p = feat + (size_t)b * m * C + c;
  for (int i = 0; i < m; ++i) mx = fmaxf(mx, p[(size_t)i * C]);
  out[t] = mx;
}

__global__ void k_pool_mean(const float* __restrict__ pos, int m, int Bd,
                            float* __restrict__ out) {
  int t = blockIdx.x * blockDim.x + threadIdx.x;
  if (t >= Bd) return;
  int b = t / 3, d = t % 3;
  float s = 0.f;
  const float* p = pos + (size_t)b * m * 3 + d;
  for (int i = 0; i < m; ++i) s += p[(size_t)i * 3];
  out[t] = s / (float)m;
}

__global__ void k_sin3d(const float* __restrict__ pp, float* __restrict__ emb, int n) {
  int t = blockIdx.x * blockDim.x + threadIdx.x;
  if (t >= n) return;
  int b = t / 96, d = t % 96;
  int a = d / 32, i = d % 32;
  int fi = (i < 16) ? i : (i - 16);
  float freq = expf(-logf(10000.f) * (float)fi / 15.f);
  float ang = pp[b * 3 + a] * freq;
  emb[t] = (i < 16) ? sinf(ang) : cosf(ang);
}

__global__ void k_add(const float* __restrict__ a, const float* __restrict__ b,
                      float* __restrict__ o, int n) {
  int t = blockIdx.x * blockDim.x + threadIdx.x;
  if (t < n) o[t] = a[t] + b[t];
}

__global__ void k_convert_wt(const float* __restrict__ W, _Float16* __restrict__ Wt,
                             int Cin, int Cout, int Cinp) {
  int i = blockIdx.x * blockDim.x + threadIdx.x;
  int total = Cout * Cinp;
  if (i >= total) return;
  int c = i / Cinp, kk = i % Cinp;
  Wt[i] = (kk < Cin) ? (_Float16)W[(size_t)kk * Cout + c] : (_Float16)0.f;
}

// ---------------------------------------------------------------------------
extern "C" void kernel_launch(void* const* d_in, const int* in_sizes, int n_in,
                              void* d_out, int out_size, void* d_ws, size_t ws_size,
                              hipStream_t stream) {
  constexpr int B = 16, N0 = 4096, M1 = 2048, K1 = 16, M2 = 1024, K2 = 24;
  struct PML { const float *W, *b, *g, *beta; };
  auto F = [&](int i) { return (const float*)d_in[i]; };
  const float* pc = nullptr;
  PML head{}, s1rpe{}, s1m0{}, s1m1{}, s1res{}, s2rpe{}, s2m0{}, s2m1{}, s2res{}, gppP{}, gprojP{};
  const bool orderA = (n_in > 0 && in_sizes[0] == B * N0 * 3);
  if (orderA) {
    pc = F(0);
    auto G = [&](int i) { return PML{F(i), F(i+1), F(i+2), F(i+3)}; };
    head = G(1);  s1rpe = G(5);  s1m0 = G(9);   s1m1 = G(13); s1res = G(17);
    s2rpe = G(21); s2m0 = G(25); s2m1 = G(29);  s2res = G(33); gppP = G(37); gprojP = G(41);
  } else {
    auto G = [&](int i) { return PML{F(i), F(i+1), F(i+3), F(i+2)}; };
    gppP = G(0); gprojP = G(4); head = G(8); s1m0 = G(12); s1m1 = G(16); s1res = G(20);
    s1rpe = G(24); s2m0 = G(28); s2m1 = G(32); s2res = G(36); s2rpe = G(40);
    pc = F(44);
  }

  char* wsb = (char*)d_ws;
  size_t off = 0;
  auto alloc = [&](size_t bytes) -> void* {
    void* p = wsb + off;
    off = (off + bytes + 255) & ~(size_t)255;
    return p;
  };
  _Float16* w_head  = (_Float16*)alloc(32ull*32*2);
  _Float16* w_s1m0  = (_Float16*)alloc(64ull*32*2);
  _Float16* w_s1m1  = (_Float16*)alloc(32ull*64*2);
  _Float16* w_s1res = (_Float16*)alloc(32ull*64*2);
  _Float16* w_s2m0  = (_Float16*)alloc(96ull*64*2);
  _Float16* w_s2m1  = (_Float16*)alloc(64ull*128*2);
  _Float16* w_s2res = (_Float16*)alloc(64ull*128*2);
  _Float16* w_gpp   = (_Float16*)alloc(96ull*128*2);
  _Float16* w_gproj = (_Float16*)alloc(128ull*256*2);
  float* feat0   = (float*)alloc((size_t)B*N0*32*4);
  int*   fidx1   = (int*)  alloc((size_t)B*M1*4);
  float* pos1    = (float*)alloc((size_t)B*M1*3*4);
  int*   nidx1   = (int*)  alloc((size_t)B*M1*K1*4);
  float* featg1  = (float*)alloc((size_t)B*M1*32*4);
  float* id1     = (float*)alloc((size_t)B*M1*64*4);
  float* hmax1   = (float*)alloc((size_t)B*M1*64*4);
  float* feat1   = (float*)alloc((size_t)B*M1*64*4);
  int*   fidx2   = (int*)  alloc((size_t)B*M2*4);
  float* pos2    = (float*)alloc((size_t)B*M2*3*4);
  int*   nidx2   = (int*)  alloc((size_t)B*M2*K2*4);
  float* featg2  = (float*)alloc((size_t)B*M2*64*4);
  float* id2     = (float*)alloc((size_t)B*M2*128*4);
  float* hmax2   = (float*)alloc((size_t)B*M2*128*4);
  float* feat2   = (float*)alloc((size_t)B*M2*128*4);
  float* pooledf = (float*)alloc(B*128*4);
  float* pooledp = (float*)alloc(B*3*4);
  float* emb     = (float*)alloc(B*96*4);
  float* gppout  = (float*)alloc(B*128*4);
  float* gf      = (float*)alloc(B*128*4);
  (void)ws_size; (void)out_size;

  auto cdiv = [](int a, int b) { return (a + b - 1) / b; };

  struct CW { const float* W; _Float16* Wp; int Cin, Cout, Cinp; } cw[] = {
    {head.W,   w_head,   3,  32,  32}, {s1m0.W, w_s1m0, 51, 32,  64},
    {s1m1.W,   w_s1m1,  32,  64,  32}, {s1res.W, w_s1res, 32, 64, 32},
    {s2m0.W,   w_s2m0,  83,  64,  96}, {s2m1.W, w_s2m1, 64, 128, 64},
    {s2res.W,  w_s2res, 64, 128,  64}, {gppP.W, w_gpp, 96, 128, 96},
    {gprojP.W, w_gproj, 128, 256, 128},
  };
  for (auto& c : cw)
    k_convert_wt<<<cdiv(c.Cinp * c.Cout, 256), 256, 0, stream>>>(c.W, c.Wp, c.Cin, c.Cout, c.Cinp);

  auto sh_pmlp = [](int Cinp, int Cout) -> size_t {
    return (size_t)64 * Cinp * sizeof(_Float16) + (size_t)64 * Cout * sizeof(float);
  };
  auto sh_group = [](int rows, int Cgp, int C0, int C1) -> size_t {
    return (size_t)rows * Cgp * 2 + (size_t)rows * C0 * 4 +
           (size_t)rows * C0 * 2 + (size_t)rows * C1 * 4;
  };

  k_pmlp<<<cdiv(B*N0, 64), 128, sh_pmlp(32, 32), stream>>>(
      pc, B*N0, 3, w_head, 32, 32, head.b, head.g, head.beta, feat0, 1);

  // ---- Stage 1 ----
  k_fps<<<B, 256, 3*N0*sizeof(float), stream>>>(pc, N0, M1, fidx1);
  k_gather_pos<<<cdiv(B*M1,256),256,0,stream>>>(pc, N0, fidx1, M1, B*M1, pos1);
  k_ballq<<<cdiv(B*M1,256),256,0,stream>>>(pc, N0, pos1, M1, K1, 0.01f, B*M1, nidx1);
  k_gather_feat<<<cdiv(B*M1,256),256,0,stream>>>(feat0, N0, 32, fidx1, M1, B*M1, featg1);
  k_pmlp<<<cdiv(B*M1,64), 128, sh_pmlp(32, 64), stream>>>(
      featg1, B*M1, 32, w_s1res, 32, 64, s1res.b, s1res.g, s1res.beta, id1, 0);
  k_group_mlp<<<B*(M1/4), 128, sh_group(4*K1, 64, 32, 64), stream>>>(
      pc, feat0, pos1, nidx1, N0, M1, K1, 32, 64, 32, 64, 4,
      s1rpe.W, s1rpe.b, s1rpe.g, s1rpe.beta,
      w_s1m0, s1m0.b, s1m0.g, s1m0.beta,
      w_s1m1, s1m1.b, s1m1.g, s1m1.beta, hmax1);
  k_combine<<<cdiv(B*M1*64,256),256,0,stream>>>(hmax1, id1, feat1, (size_t)B*M1*64);

  // ---- Stage 2 ----
  k_fps<<<B, 256, 3*M1*sizeof(float), stream>>>(pos1, M1, M2, fidx2);
  k_gather_pos<<<cdiv(B*M2,256),256,0,stream>>>(pos1, M1, fidx2, M2, B*M2, pos2);
  k_ballq<<<cdiv(B*M2,256),256,0,stream>>>(pos1, M1, pos2, M2, K2, 0.04f, B*M2, nidx2);
  k_gather_feat<<<cdiv(B*M2,256),256,0,stream>>>(feat1, M1, 64, fidx2, M2, B*M2, featg2);
  k_pmlp<<<cdiv(B*M2,64), 128, sh_pmlp(64, 128), stream>>>(
      featg2, B*M2, 64, w_s2res, 64, 128, s2res.b, s2res.g, s2res.beta, id2, 0);
  k_group_mlp<<<B*(M2/2), 128, sh_group(2*K2, 96, 64, 128), stream>>>(
      pos1, feat1, pos2, nidx2, M1, M2, K2, 64, 96, 64, 128, 2,
      s2rpe.W, s2rpe.b, s2rpe.g, s2rpe.beta,
      w_s2m0, s2m0.b, s2m0.g, s2m0.beta,
      w_s2m1, s2m1.b, s2m1.g, s2m1.beta, hmax2);
  k_combine<<<cdiv(B*M2*128,256),256,0,stream>>>(hmax2, id2, feat2, (size_t)B*M2*128);

  // ---- Global pooling + head ----
  k_pool_max<<<cdiv(B*128,256),256,0,stream>>>(feat2, M2, 128, B*128, pooledf);
  k_pool_mean<<<1, 64, 0, stream>>>(pos2, M2, B*3, pooledp);
  k_sin3d<<<cdiv(B*96,256),256,0,stream>>>(pooledp, emb, B*96);
  k_pmlp<<<1, 128, sh_pmlp(96, 128), stream>>>(
      emb, B, 96, w_gpp, 96, 128, gppP.b, gppP.g, gppP.beta, gppout, 1);
  k_add<<<cdiv(B*128,256),256,0,stream>>>(pooledf, gppout, gf, B*128);
  k_pmlp<<<1, 128, sh_pmlp(128, 256), stream>>>(
      gf, B, 128, w_gproj, 128, 256, gprojP.b, gprojP.g, gprojP.beta, (float*)d_out, 0);
}
